// QKVAttention_240518168648
// MI455X (gfx1250) — matrix operand order repaired and checked by reading, not compiled, not run on hardware
//
#include <hip/hip_runtime.h>
#include <hip/hip_bf16.h>

// ---------------------------------------------------------------------------
// Flash-attention for qkv[4,1536,2048] fp32, H=8, C=64, T=2048.
// Transposed-score formulation: S^T = K^T Q so the softmax axis lies on the
// register (row) axis of the WMMA D fragment -> in-lane reductions + one
// cross-half shuffle; P feeds the second GEMM (O^T = V P^T) as a B operand
// with only a 4-dword cross-half swap. bf16 WMMA, fp32 softmax/accumulate.
// Workgroup: 256 threads = 8 wave32; each wave owns 16 output columns (t).
// Double-buffered K/V LDS tiles; packed bf16 staging stores.
// ---------------------------------------------------------------------------

typedef __attribute__((ext_vector_type(16))) __bf16 v16bf;
typedef __attribute__((ext_vector_type(8)))  __bf16 bf16x8;
typedef __attribute__((ext_vector_type(4)))  __bf16 bf16x4;
typedef __attribute__((ext_vector_type(8)))  float  v8f;

#define T_LEN    2048
#define CH       64         // head dim
#define QTILE    128        // query columns per workgroup (16 per wave)
#define SBLK     64         // key/value rows per iteration
#define NSB      (T_LEN / SBLK)
#define NTHREADS 256

// LDS row strides (bf16 elems): rows 16B-aligned, 16-lane column walks hit
// distinct banks (step 44 resp. 36 mod 64).
#define QT_STRIDE 88        // qT[t][c]   (B-operand source for scores)
#define KT_STRIDE 88        // kT[s][c]   (A-operand source for scores)
#define V_STRIDE  72        // vB[c][s]   (A-operand source for PV)

static __device__ __forceinline__ v16bf cat8(bf16x8 lo, bf16x8 hi) {
    return __builtin_shufflevector(lo, hi, 0,1,2,3,4,5,6,7,8,9,10,11,12,13,14,15);
}
static __device__ __forceinline__ v8f wmma_bf16(v16bf a, v16bf b, v8f c) {
    return __builtin_amdgcn_wmma_f32_16x16x32_bf16(
        false, a, false, b, (short)0, c, false, false);
}
static __device__ __forceinline__ unsigned pk2(float x, float y) {
    union { __bf16 h[2]; unsigned u; } t;
    t.h[0] = (__bf16)x; t.h[1] = (__bf16)y;
    return t.u;
}

// Stage one s-block: K -> kT[s][c] (transposed, c-paired b32 stores),
// V -> vB[c][s] (natural, b64 stores), both bf16.
static __device__ __forceinline__ void stageKV(const float* __restrict__ qkv,
                                               size_t kBase, size_t vBase, int s0,
                                               __bf16* kTb, __bf16* vBb, int tid)
{
    for (int idx = tid; idx < (CH / 2) * (SBLK / 4); idx += NTHREADS) {
        const int cp = idx >> 4;           // c-pair index 0..31
        const int s4 = (idx & 15) << 2;    // s-quad base
        const size_t go = (size_t)(2 * cp) * T_LEN + s0 + s4;
        const float4 k0 = *(const float4*)&qkv[kBase + go];
        const float4 k1 = *(const float4*)&qkv[kBase + go + T_LEN];
        const float4 v0 = *(const float4*)&qkv[vBase + go];
        const float4 v1 = *(const float4*)&qkv[vBase + go + T_LEN];
        // kT: column pair (2cp, 2cp+1) of rows s4..s4+3 -> packed b32 stores
        unsigned* kc = (unsigned*)&kTb[s4 * KT_STRIDE + 2 * cp];
        kc[0 * (KT_STRIDE / 2)] = pk2(k0.x, k1.x);
        kc[1 * (KT_STRIDE / 2)] = pk2(k0.y, k1.y);
        kc[2 * (KT_STRIDE / 2)] = pk2(k0.z, k1.z);
        kc[3 * (KT_STRIDE / 2)] = pk2(k0.w, k1.w);
        // vB: rows 2cp, 2cp+1, contiguous s-quads -> b64 stores
        const bf16x4 a0 = { (__bf16)v0.x, (__bf16)v0.y, (__bf16)v0.z, (__bf16)v0.w };
        const bf16x4 a1 = { (__bf16)v1.x, (__bf16)v1.y, (__bf16)v1.z, (__bf16)v1.w };
        *(bf16x4*)&vBb[(2 * cp)     * V_STRIDE + s4] = a0;
        *(bf16x4*)&vBb[(2 * cp + 1) * V_STRIDE + s4] = a1;
    }
}

__global__ __launch_bounds__(NTHREADS)
void QKVAttention_240518168648_kernel(const float* __restrict__ qkv,
                                      float* __restrict__ out)
{
    __shared__ __align__(16) __bf16 qT[QTILE * QT_STRIDE];
    __shared__ __align__(16) __bf16 kT[2][SBLK * KT_STRIDE];
    __shared__ __align__(16) __bf16 vB[2][CH * V_STRIDE];

    const int tid  = threadIdx.x;
    const int lane = tid & 31;
    const int wave = tid >> 5;
    const int col  = lane & 15;        // lane's N index (= t within subtile)
    const int lh   = (lane >> 4) & 1;  // ISA fragment lane-half

    const int bh = blockIdx.y;         // b*8 + h
    const int b  = bh >> 3;
    const int h  = bh & 7;
    const int q0 = blockIdx.x * QTILE;

    const size_t qBase = ((size_t)b * 1536 + (size_t)h * CH) * T_LEN;
    const size_t kBase = qBase + (size_t)512  * T_LEN;
    const size_t vBase = qBase + (size_t)1024 * T_LEN;
    const float  scale = 0.125f;       // (1/ch^0.25)^2 folded into Q

    // ---- stage Q tile once: qT[t][c] = q[c][q0+t] * scale (c-paired) ----
    for (int idx = tid; idx < (CH / 2) * (QTILE / 4); idx += NTHREADS) {
        const int cp = idx >> 5;           // c-pair 0..31
        const int t4 = (idx & 31) << 2;    // t-quad base
        const size_t go = (size_t)(2 * cp) * T_LEN + q0 + t4;
        const float4 x0 = *(const float4*)&qkv[qBase + go];
        const float4 x1 = *(const float4*)&qkv[qBase + go + T_LEN];
        unsigned* qc = (unsigned*)&qT[t4 * QT_STRIDE + 2 * cp];
        qc[0 * (QT_STRIDE / 2)] = pk2(x0.x * scale, x1.x * scale);
        qc[1 * (QT_STRIDE / 2)] = pk2(x0.y * scale, x1.y * scale);
        qc[2 * (QT_STRIDE / 2)] = pk2(x0.z * scale, x1.z * scale);
        qc[3 * (QT_STRIDE / 2)] = pk2(x0.w * scale, x1.w * scale);
    }
    stageKV(qkv, kBase, vBase, 0, kT[0], vB[0], tid);
    __syncthreads();

    // ---- per-wave Q B-fragments (lane n = t, elem i -> K = i+16*lh, contiguous k order)
    v16bf Bq[2];
    {
        const __bf16* rowp = &qT[(wave * 16 + col) * QT_STRIDE];
        #pragma unroll
        for (int c2 = 0; c2 < 2; ++c2) {
            const bf16x8 lo = *(const bf16x8*)(rowp + 32 * c2 + 16 * lh);
            const bf16x8 hi = *(const bf16x8*)(rowp + 32 * c2 + 8 + 16 * lh);
            Bq[c2] = cat8(lo, hi);
        }
    }

    // per-lane online-softmax state for column t = q0 + wave*16 + col
    float m = -INFINITY, l = 0.0f;
    v8f O[4] = {};                     // O^T[c][t]: VGPR r <-> c = 16*j + r + 8*lh

    for (int sb = 0; sb < NSB; ++sb) {
        const __bf16* kTb = kT[sb & 1];
        const __bf16* vBb = vB[sb & 1];

        // stage next tile into the other buffer while this one is consumed
        if (sb + 1 < NSB)
            stageKV(qkv, kBase, vBase, (sb + 1) * SBLK,
                    kT[(sb + 1) & 1], vB[(sb + 1) & 1], tid);
        if (sb + 2 < NSB) {
            const int    pc = tid & 63;
            const size_t pb_ = (tid < 128) ? kBase : vBase;
            __builtin_prefetch(&qkv[pb_ + (size_t)pc * T_LEN + (sb + 2) * SBLK], 0, 3);
        }

        // ---- scores S^T[s][t], 4 s-subtiles of 16; A = kT rows, B = Bq ----
        v8f S[4];
        #pragma unroll
        for (int st = 0; st < 4; ++st) {
            const __bf16* krow = &kTb[(16 * st + col) * KT_STRIDE];
            v8f z = {};
            #pragma unroll
            for (int c2 = 0; c2 < 2; ++c2) {
                const bf16x8 lo = *(const bf16x8*)(krow + 32 * c2 + 16 * lh);
                const bf16x8 hi = *(const bf16x8*)(krow + 32 * c2 + 8 + 16 * lh);
                z = wmma_bf16(cat8(lo, hi), Bq[c2], z);
            }
            S[st] = z;
        }

        // ---- online softmax along registers: in-lane + one cross-half op ----
        float tmax = -INFINITY;
        #pragma unroll
        for (int st = 0; st < 4; ++st)
            #pragma unroll
            for (int r = 0; r < 8; ++r) tmax = fmaxf(tmax, S[st][r]);
        tmax = fmaxf(tmax, __shfl_xor(tmax, 16, 32));
        const float mnew  = fmaxf(m, tmax);
        const float alpha = __expf(m - mnew);
        m = mnew;

        float p[4][8];
        float sum = 0.0f;
        #pragma unroll
        for (int st = 0; st < 4; ++st)
            #pragma unroll
            for (int r = 0; r < 8; ++r) {
                p[st][r] = __expf(S[st][r] - mnew);
                sum += p[st][r];
            }
        sum += __shfl_xor(sum, 16, 32);
        l = l * alpha + sum;
        #pragma unroll
        for (int j = 0; j < 4; ++j) O[j] *= alpha;

        // ---- O^T += V * P^T, two K=32 steps over the 64 s rows ----
        #pragma unroll
        for (int kk = 0; kk < 2; ++kk) {
            // P B-fragment: lane col = t, elem i -> K(s32) = i + 16*lh.
            // subtile a = S[2kk] (s32 = r+8*lh), b = S[2kk+1] (s32 = 16+r+8*lh):
            // half0 needs {own a, partner a}; half1 needs {partner b, own b}.
            unsigned pa[4], pb[4];
            #pragma unroll
            for (int d = 0; d < 4; ++d) {
                pa[d] = pk2(p[2 * kk][2 * d],     p[2 * kk][2 * d + 1]);
                pb[d] = pk2(p[2 * kk + 1][2 * d], p[2 * kk + 1][2 * d + 1]);
            }
            union { v16bf v; unsigned u[8]; } P;
            #pragma unroll
            for (int d = 0; d < 4; ++d) {
                const unsigned snd = lh ? pa[d] : pb[d];
                const unsigned rcv = __shfl_xor(snd, 16, 32);
                P.u[d]     = lh ? rcv   : pa[d];
                P.u[d + 4] = lh ? pb[d] : rcv;
            }
            #pragma unroll
            for (int j = 0; j < 4; ++j) {
                const __bf16* vrow = &vBb[(16 * j + col) * V_STRIDE + 32 * kk];
                const bf16x8 lo = *(const bf16x8*)(vrow + 16 * lh);
                const bf16x8 hi = *(const bf16x8*)(vrow + 8 + 16 * lh);
                O[j] = wmma_bf16(cat8(lo, hi), P.v, O[j]);
            }
        }

        __syncthreads();   // next buffer fully staged / this one fully consumed
    }

    // ---- epilogue: coalesced stores, out[b, h*64 + c, t] = O^T / l ----
    const float inv = 1.0f / l;
    const size_t outBase = ((size_t)b * 512 + (size_t)h * CH) * T_LEN;
    const int tcol = q0 + wave * 16 + col;
    #pragma unroll
    for (int j = 0; j < 4; ++j)
        #pragma unroll
        for (int r = 0; r < 8; ++r) {
            const int c = 16 * j + r + 8 * lh;
            out[outBase + (size_t)c * T_LEN + tcol] = O[j][r] * inv;
        }
}

extern "C" void kernel_launch(void* const* d_in, const int* in_sizes, int n_in,
                              void* d_out, int out_size, void* d_ws, size_t ws_size,
                              hipStream_t stream) {
    (void)in_sizes; (void)n_in; (void)out_size; (void)d_ws; (void)ws_size;
    const float* qkv = (const float*)d_in[0];
    float* out = (float*)d_out;
    dim3 grid(T_LEN / QTILE, 32);     // 16 q-tiles x (4 batches * 8 heads)
    QKVAttention_240518168648_kernel<<<grid, NTHREADS, 0, stream>>>(qkv, out);
}
